// Attention_50268297232602
// MI455X (gfx1250) — compile-verified, matrix operand order checked by reading, not run on hardware
//
#include <hip/hip_runtime.h>

typedef __attribute__((ext_vector_type(16))) __bf16 v16bf;
typedef __attribute__((ext_vector_type(8)))  __bf16 v8bf;
typedef __attribute__((ext_vector_type(8)))  float  v8f;

#define S_TOK  2048
#define DMODEL 1024
#define NHEAD  16
#define HDIM   64
#define GF     8
#define GH     16

// LDS row padding (elements) to rotate banks: +16B per row (4-bank shift)
#define APAD   8
#define AROW   (DMODEL + APAD)

union FragBF {
    v16bf v;
    struct { v8bf lo, hi; } h;
};

__device__ __forceinline__ v8f wmma_bf16(v16bf a, v16bf b, v8f c) {
    // D = A(16x32 bf16) * B(32x16 bf16) + C(16x16 f32)
    return __builtin_amdgcn_wmma_f32_16x16x32_bf16(false, a, false, b,
                                                   (short)0, c, false, false);
}

// workgroup-relative LDS byte offset of a shared-memory pointer
typedef __attribute__((address_space(3))) uint8_t lds_u8;
__device__ __forceinline__ uint32_t lds_offset(const void* p) {
    return (uint32_t)(uintptr_t)(lds_u8*)p;
}

__device__ __forceinline__ void async_copy_b128(uint32_t lds_byte_off,
                                                const void* gaddr) {
    asm volatile("global_load_async_to_lds_b128 %0, %1, off"
                 :: "v"(lds_byte_off), "v"((uint64_t)(uintptr_t)gaddr)
                 : "memory");
}

__device__ __forceinline__ void wait_async0() {
    asm volatile("s_wait_asynccnt 0x0" ::: "memory");
}

__device__ __forceinline__ FragBF load_frag(const __bf16* row, int kb, int half) {
    FragBF f;
    f.h.lo = *(const v8bf*)(row + kb + half * 8);
    f.h.hi = *(const v8bf*)(row + kb + 16 + half * 8);
    return f;
}

__device__ __forceinline__ void load_b4(FragBF b[4], const __bf16* const wrow[4],
                                        int kb, int half) {
#pragma unroll
    for (int t = 0; t < 4; ++t) b[t] = load_frag(wrow[t], kb, half);
}

__device__ __forceinline__ void wmma4(v8f acc[4], const FragBF& a, const FragBF b[4]) {
#pragma unroll
    for (int t = 0; t < 4; ++t) acc[t] = wmma_bf16(a.v, b[t].v, acc[t]);
}

// ---------------------------------------------------------------------------
// fp32 -> bf16 conversion (hidden states)
// ---------------------------------------------------------------------------
__global__ void k_cvt_f32_bf16(const float* __restrict__ in,
                               __bf16* __restrict__ out, int n) {
    int i = blockIdx.x * blockDim.x + threadIdx.x;
    if (i < n) out[i] = (__bf16)in[i];
}

// W [K=1024][N=1024] fp32 row-major -> Wt [N][K] bf16 (B-fragment friendly)
__global__ void k_cvt_transpose_w(const float* __restrict__ W,
                                  __bf16* __restrict__ Wt) {
    int i = blockIdx.x * blockDim.x + threadIdx.x;   // over D*D
    int n = i >> 10;
    int k = i & (DMODEL - 1);
    Wt[i] = (__bf16)W[k * DMODEL + n];
}

// ---------------------------------------------------------------------------
// Shared GEMM core: 4 waves per workgroup, M=16 x N=256 tile.
// A block (16 x 1024 bf16, contiguous 32KB) async-staged to LDS once and
// shared by all waves; B (Wt, [N][K]) streamed from global with register
// double-buffering so WMMAs overlap the next chunk's loads.
// ---------------------------------------------------------------------------
__device__ __forceinline__ void gemm_core(const __bf16* __restrict__ X,
                                          const __bf16* __restrict__ Wt,
                                          __bf16 (*sA)[AROW],
                                          v8f acc[4],
                                          int m0, int n0w,
                                          int half, int r) {
    const int tid = threadIdx.x;

    // ---- async stage A block: rows m0..m0+15, all K (32KB contiguous) ----
    {
        const uint8_t* gsrc = (const uint8_t*)(X + (size_t)m0 * DMODEL);
        const uint32_t lbase = lds_offset(&sA[0][0]);
        for (int c = tid; c < (16 * DMODEL * 2) / 16; c += 128) {
            const int row = c >> 7;          // 128 x 16B chunks per row
            const int col = c & 127;
            async_copy_b128(lbase + (uint32_t)(row * (AROW * 2) + col * 16),
                            gsrc + (size_t)row * (DMODEL * 2) + col * 16);
        }
    }

    const __bf16* wrow[4];
#pragma unroll
    for (int t = 0; t < 4; ++t)
        wrow[t] = Wt + (size_t)(n0w + t * 16 + r) * DMODEL;

    FragBF b0[4], b1[4];
    load_b4(b0, wrow, 0, half);              // overlaps with async staging

    wait_async0();
    __syncthreads();

    const __bf16* arow = &sA[r][0];

    for (int kc = 0; kc < DMODEL / 32; kc += 2) {
        const int kb0 = kc * 32;
        load_b4(b1, wrow, kb0 + 32, half);
        FragBF a = load_frag(arow, kb0, half);
        wmma4(acc, a, b0);

        if (kc + 2 < DMODEL / 32) load_b4(b0, wrow, kb0 + 64, half);
        a = load_frag(arow, kb0 + 32, half);
        wmma4(acc, a, b1);
    }
}

// mode: 0=Q (scale 1/8, row-major), 1=K (row-major), 2=V (transposed Vt[n][m])
__global__ void __launch_bounds__(128)
k_gemm_qkv(const __bf16* __restrict__ X, const __bf16* __restrict__ Wt,
           __bf16* __restrict__ out, int mode) {
    __shared__ __bf16 sA[16][AROW];

    const int lane = threadIdx.x & 31;
    const int wave = threadIdx.x >> 5;
    const int half = lane >> 4;
    const int r    = lane & 15;
    const int m0   = blockIdx.y * 16;
    const int n0w  = blockIdx.x * 256 + wave * 64;

    v8f acc[4] = {};
    gemm_core(X, Wt, sA, acc, m0, n0w, half, r);

    const float scale = (mode == 0) ? 0.125f : 1.0f;   // 1/sqrt(64) exact
#pragma unroll
    for (int t = 0; t < 4; ++t) {
        const int col = n0w + t * 16 + r;              // C layout: lane = n
#pragma unroll
        for (int j = 0; j < 8; ++j) {
            const int row = m0 + j + 8 * half;         // vgpr j = m
            const float v = acc[t][j] * scale;
            if (mode == 2) out[(size_t)col * S_TOK + row] = (__bf16)v;
            else           out[(size_t)row * DMODEL + col] = (__bf16)v;
        }
    }
}

__global__ void __launch_bounds__(128)
k_gemm_out(const __bf16* __restrict__ X, const __bf16* __restrict__ Wt,
           const float* __restrict__ bias, float* __restrict__ out) {
    __shared__ __bf16 sA[16][AROW];

    const int lane = threadIdx.x & 31;
    const int wave = threadIdx.x >> 5;
    const int half = lane >> 4;
    const int r    = lane & 15;
    const int m0   = blockIdx.y * 16;
    const int n0w  = blockIdx.x * 256 + wave * 64;

    v8f acc[4] = {};
    gemm_core(X, Wt, sA, acc, m0, n0w, half, r);

#pragma unroll
    for (int t = 0; t < 4; ++t) {
        const int col = n0w + t * 16 + r;
        const float bv = bias[col];
#pragma unroll
        for (int j = 0; j < 8; ++j) {
            const int row = m0 + j + 8 * half;
            out[(size_t)row * DMODEL + col] = acc[t][j] + bv;
        }
    }
}

// ---------------------------------------------------------------------------
// Block-sparse windowed attention. One wave per (q-block of 16, head).
// q-block = fixed (f,h), w = 0..15. Valid kv-blocks: |df|<=1, |dh|<=2
// (<=15 blocks). Elementwise mask within a block pair: |w_q - w_k| <= 2.
// ---------------------------------------------------------------------------
#define SPAD  (256 + 1)    // fp32 rows: +4B  -> 1-bank rotation per row
#define PPAD  (256 + 8)    // bf16 rows: +16B -> 4-bank rotation per row

__global__ void __launch_bounds__(32)
k_attn(const __bf16* __restrict__ Qb, const __bf16* __restrict__ Kb,
       const __bf16* __restrict__ Vt, __bf16* __restrict__ Ab) {
    __shared__ float  sbuf[16][SPAD];  // scores, fp32
    __shared__ __bf16 pbuf[16][PPAD];  // softmax probs, bf16 (zero padded)
    __shared__ float  rinv[16];
    __shared__ int    kblist[16];

    const int lane = threadIdx.x;
    const int half = lane >> 4;
    const int r    = lane & 15;
    const int qb   = blockIdx.x;       // 0..127
    const int head = blockIdx.y;       // 0..15

    const int f  = qb >> 4;
    const int hq = qb & 15;
    const int f0 = (f - 1 < 0) ? 0 : f - 1;
    const int f1 = (f + 1 > GF - 1) ? GF - 1 : f + 1;
    const int h0 = (hq - 2 < 0) ? 0 : hq - 2;
    const int h1 = (hq + 2 > GH - 1) ? GH - 1 : hq + 2;
    const int nblk = (f1 - f0 + 1) * (h1 - h0 + 1);

    if (lane == 0) {
        int j = 0;
        for (int ff = f0; ff <= f1; ++ff)
            for (int hh = h0; hh <= h1; ++hh)
                kblist[j++] = ff * 16 + hh;
    }
    // zero P buffer (padding for odd kv-block counts contributes nothing)
    int4* p4 = (int4*)&pbuf[0][0];
    for (int i = lane; i < (int)(16 * PPAD * 2) / 16; i += 32)
        p4[i] = make_int4(0, 0, 0, 0);
    __syncthreads();

    const int q0 = qb * 16;
    const __bf16* qrow = Qb + (size_t)(q0 + r) * DMODEL + head * HDIM;
    FragBF qa[2];
#pragma unroll
    for (int c = 0; c < 2; ++c) qa[c] = load_frag(qrow, c * 32, half);

    // ---- Phase 1: score tiles S = (Q*scale) K^T, masked on |w_q - w_k| ----
    for (int j = 0; j < nblk; ++j) {
        const int kv0 = kblist[j] * 16;
        const __bf16* krow = Kb + (size_t)(kv0 + r) * DMODEL + head * HDIM;
        FragBF kf[2];
#pragma unroll
        for (int c = 0; c < 2; ++c) kf[c] = load_frag(krow, c * 32, half);
        v8f s = {};
        s = wmma_bf16(qa[0].v, kf[0].v, s);
        s = wmma_bf16(qa[1].v, kf[1].v, s);
#pragma unroll
        for (int jj = 0; jj < 8; ++jj) {
            const int m = jj + 8 * half;   // query w
            const int n = r;               // kv w
            int d = m - n; d = d < 0 ? -d : d;
            float v = (d > 2) ? -1e9f : s[jj];
            sbuf[m][j * 16 + n] = v;
        }
    }
    __syncthreads();

    // ---- Phase 2: softmax, two lanes per row (halves), shfl_xor combine ----
    const int Lh = nblk * 8;
    const int c0 = half * Lh;
    float mx = -1e30f;
    for (int i = 0; i < Lh; ++i) mx = fmaxf(mx, sbuf[r][c0 + i]);
    mx = fmaxf(mx, __shfl_xor(mx, 16, 32));
    float sum = 0.f;
    for (int i = 0; i < Lh; ++i) {
        float p = __expf(sbuf[r][c0 + i] - mx);
        sum += p;
        pbuf[r][c0 + i] = (__bf16)p;
    }
    sum += __shfl_xor(sum, 16, 32);
    if (half == 0) rinv[r] = 1.0f / sum;
    __syncthreads();

    // ---- Phase 3: O = P * V (V pre-transposed: Vt[dim][token]) ----
    const int nkc = (nblk + 1) >> 1;       // 32-column P chunks
    v8f o[4] = {};
    for (int kc = 0; kc < nkc; ++kc) {
        const int kb0 = kc * 32;
        FragBF pa = load_frag(&pbuf[r][0], kb0, half);
        const int jA = 2 * kc;
        const int jB = (jA + 1 < nblk) ? jA + 1 : jA;   // clamp (pad P is 0)
        const int tokA = kblist[jA] * 16 + half * 8;
        const int tokB = kblist[jB] * 16 + half * 8;
#pragma unroll
        for (int t = 0; t < 4; ++t) {
            const __bf16* vcol = Vt + (size_t)(head * HDIM + t * 16 + r) * S_TOK;
            FragBF vb;
            vb.h.lo = *(const v8bf*)(vcol + tokA);
            vb.h.hi = *(const v8bf*)(vcol + tokB);
            o[t] = wmma_bf16(pa.v, vb.v, o[t]);
        }
    }

#pragma unroll
    for (int t = 0; t < 4; ++t) {
#pragma unroll
        for (int j = 0; j < 8; ++j) {
            const int m = j + 8 * half;
            const float v = o[t][j] * rinv[m];
            Ab[(size_t)(q0 + m) * DMODEL + head * HDIM + t * 16 + r] = (__bf16)v;
        }
    }
}

// ---------------------------------------------------------------------------
extern "C" void kernel_launch(void* const* d_in, const int* in_sizes, int n_in,
                              void* d_out, int out_size, void* d_ws, size_t ws_size,
                              hipStream_t stream) {
    (void)in_sizes; (void)n_in; (void)out_size; (void)ws_size;

    const float* H  = (const float*)d_in[0];
    const float* Wq = (const float*)d_in[1];
    const float* Wk = (const float*)d_in[2];
    const float* Wv = (const float*)d_in[3];
    const float* Wo = (const float*)d_in[4];
    const float* bo = (const float*)d_in[5];
    float* out = (float*)d_out;

    char* ws = (char*)d_ws;
    size_t off = 0;
    auto alloc = [&](size_t bytes) {
        void* p = ws + off;
        off += (bytes + 255) & ~(size_t)255;
        return p;
    };
    __bf16* Hb  = (__bf16*)alloc((size_t)S_TOK * DMODEL * 2);
    __bf16* Wqt = (__bf16*)alloc((size_t)DMODEL * DMODEL * 2);
    __bf16* Wkt = (__bf16*)alloc((size_t)DMODEL * DMODEL * 2);
    __bf16* Wvt = (__bf16*)alloc((size_t)DMODEL * DMODEL * 2);
    __bf16* Wot = (__bf16*)alloc((size_t)DMODEL * DMODEL * 2);
    __bf16* Qb  = (__bf16*)alloc((size_t)S_TOK * DMODEL * 2);
    __bf16* Kbm = (__bf16*)alloc((size_t)S_TOK * DMODEL * 2);
    __bf16* Vt  = (__bf16*)alloc((size_t)DMODEL * S_TOK * 2);
    __bf16* Ab  = (__bf16*)alloc((size_t)S_TOK * DMODEL * 2);

    // 1) precision conversion + weight transpose (bf16, B-fragment layout)
    k_cvt_f32_bf16<<<(S_TOK * DMODEL) / 256, 256, 0, stream>>>(H, Hb, S_TOK * DMODEL);
    const int wblk = (DMODEL * DMODEL) / 256;
    k_cvt_transpose_w<<<wblk, 256, 0, stream>>>(Wq, Wqt);
    k_cvt_transpose_w<<<wblk, 256, 0, stream>>>(Wk, Wkt);
    k_cvt_transpose_w<<<wblk, 256, 0, stream>>>(Wv, Wvt);
    k_cvt_transpose_w<<<wblk, 256, 0, stream>>>(Wo, Wot);

    // 2) QKV projections (Q pre-scaled by 1/8; V stored transposed)
    dim3 gg(DMODEL / 256, S_TOK / 16);
    k_gemm_qkv<<<gg, 128, 0, stream>>>(Hb, Wqt, Qb, 0);
    k_gemm_qkv<<<gg, 128, 0, stream>>>(Hb, Wkt, Kbm, 1);
    k_gemm_qkv<<<gg, 128, 0, stream>>>(Hb, Wvt, Vt, 2);

    // 3) block-sparse windowed attention
    k_attn<<<dim3(S_TOK / 16, NHEAD), 32, 0, stream>>>(Qb, Kbm, Vt, Ab);

    // 4) output projection + bias (fp32 out)
    k_gemm_out<<<gg, 128, 0, stream>>>(Ab, Wot, bo, out);
}
